// DTFN_38405597561803
// MI455X (gfx1250) — matrix operand order repaired and checked by reading, not compiled, not run on hardware
//
#include <hip/hip_runtime.h>
#include <hip/hip_bf16.h>

// ---------------- types ----------------
typedef __bf16 bf16_t;
typedef __attribute__((ext_vector_type(16))) __bf16 v16bf;
typedef __attribute__((ext_vector_type(8)))  __bf16 v8bf;
typedef __attribute__((ext_vector_type(8)))  float  v8f;
typedef int v4i_vs __attribute__((vector_size(16)));   // matches async-LDS builtin param type

#define BB 4
#define SS 2048
#define DD 128
#define VV 32000
#define KITER 3
#define EPSF 1e-6f
#define LDS_STRIDE 136   // 128 + 8 pad: 272B row stride -> conflict-free ds_load_b128

// Async global->LDS copy path (gfx1250 GLOBAL_LOAD_ASYNC_TO_LDS_B128, ASYNCcnt)
#if defined(__has_builtin)
# if __has_builtin(__builtin_amdgcn_global_load_async_to_lds_b128)
#  define USE_ASYNC_LDS 1
# endif
#endif

// ---------------- device helpers ----------------
__device__ __forceinline__ float softplus_f(float x) {
    return (x > 15.f) ? x : log1pf(__expf(x));
}

__device__ __forceinline__ v8f wmma_bf16(v16bf a, v16bf b, v8f c) {
    return __builtin_amdgcn_wmma_f32_16x16x32_bf16(
        /*neg_a=*/false, a, /*neg_b=*/false, b,
        /*c_mod=*/(short)0, c, /*reuse_a=*/false, /*reuse_b=*/false);
}

// A-matrix 16x32 bf16 fragment from one row pointer (row = lane&15 chosen by caller).
// Lanes 0-15: elems 0..7 = K kbase+0..7, elems 8..15 = K kbase+16..23
// Lanes 16-31: K offset by +8 in each half.
__device__ __forceinline__ v16bf load_a_frag_rowptr(const bf16_t* rowp, int kbase, int lane) {
    int h = (lane >> 4) & 1;
    const bf16_t* p0 = rowp + kbase + h * 8;
    v8bf a0 = *(const v8bf*)p0;
    v8bf a1 = *(const v8bf*)(p0 + 16);
    v16bf r;
#pragma unroll
    for (int i = 0; i < 8; ++i) { r[i] = a0[i]; r[i + 8] = a1[i]; }
    return r;
}

// Same but from an f32 row (convert to bf16 in regs).
__device__ __forceinline__ v16bf load_a_frag_f32row(const float* rowp, int kbase, int lane) {
    int h = (lane >> 4) & 1;
    const float* p0 = rowp + kbase + h * 8;
    v16bf r;
#pragma unroll
    for (int i = 0; i < 8; ++i) { r[i] = (bf16_t)p0[i]; r[i + 8] = (bf16_t)p0[16 + i]; }
    return r;
}

// B-matrix 32x16 bf16 fragment from global. W layout: W[n*DD + k].
// Lanes 0-15: N = n0+lane, K = kbase+0..7,16..23 ; lanes 16-31: same N, K = kbase+8..15,24..31.
__device__ __forceinline__ v16bf load_b_frag(const bf16_t* W, int n0, int kbase, int lane) {
    int n = n0 + (lane & 15);
    int k = kbase + ((lane >> 4) << 3);
    const bf16_t* p = W + (size_t)n * DD + k;
    v8bf b0 = *(const v8bf*)p;
    v8bf b1 = *(const v8bf*)(p + 16);
    v16bf r;
#pragma unroll
    for (int i = 0; i < 8; ++i) { r[i] = b0[i]; r[i + 8] = b1[i]; }
    return r;
}

// B fragment from an LDS tile with padded row stride.
__device__ __forceinline__ v16bf load_b_frag_lds(const bf16_t* tile, int n0, int kbase, int lane) {
    int n = n0 + (lane & 15);
    int k = kbase + ((lane >> 4) << 3);
    const bf16_t* p = tile + n * LDS_STRIDE + k;
    v8bf b0 = *(const v8bf*)p;
    v8bf b1 = *(const v8bf*)(p + 16);
    v16bf r;
#pragma unroll
    for (int i = 0; i < 8; ++i) { r[i] = b0[i]; r[i + 8] = b1[i]; }
    return r;
}

// ---------------- conversion kernels ----------------
__global__ void k_cvt_bf16(const float* __restrict__ src, bf16_t* __restrict__ dst, int n) {
    int i = blockIdx.x * blockDim.x + threadIdx.x;
    if (i < n) dst[i] = (bf16_t)src[i];
}

__global__ void k_split_wf1(const float* __restrict__ wf1,
                            bf16_t* __restrict__ wl, bf16_t* __restrict__ wr) {
    int i = blockIdx.x * blockDim.x + threadIdx.x;
    if (i < DD * DD) {
        int n = i >> 7, d = i & 127;
        wl[i] = (bf16_t)wf1[n * 2 * DD + d];
        wr[i] = (bf16_t)wf1[n * 2 * DD + DD + d];
    }
}

// ---------------- embed + mass: m = softplus(emb[x] @ w_mass^T + b) + eps ----------------
// one wave per 16-row tile; 512 blocks x 32 threads
__global__ void k_embed_mass(const long long* __restrict__ x, const float* __restrict__ emb,
                             const bf16_t* __restrict__ wmass_bf, const float* __restrict__ b_mass,
                             float* __restrict__ m, bf16_t* __restrict__ m_bf) {
    int lane = threadIdx.x & 31;
    int r0 = blockIdx.x * 16;
    long long tok = x[r0 + (lane & 15)];
    const float* hrow = emb + (size_t)tok * DD;

    v8f acc[8];
#pragma unroll
    for (int n = 0; n < 8; ++n)
#pragma unroll
        for (int i = 0; i < 8; ++i) acc[n][i] = 0.f;

#pragma unroll
    for (int k = 0; k < 4; ++k) {
        v16bf a = load_a_frag_f32row(hrow, k * 32, lane);
#pragma unroll
        for (int n = 0; n < 8; ++n) {
            v16bf b = load_b_frag(wmass_bf, n * 16, k * 32, lane);
            acc[n] = wmma_bf16(a, b, acc[n]);
        }
    }

    int ncol = lane & 15;
    int rbase = r0 + ((lane >> 4) << 3);
#pragma unroll
    for (int n = 0; n < 8; ++n) {
        int col = n * 16 + ncol;
        float bias = b_mass[col];
#pragma unroll
        for (int v = 0; v < 8; ++v) {
            float val = softplus_f(acc[n][v] + bias) + EPSF;
            size_t idx = (size_t)(rbase + v) * DD + col;
            m[idx] = val;
            m_bf[idx] = (bf16_t)val;
        }
    }
}

// ---------------- F step: F = softplus(tanh(L@Wl^T + R@Wr^T + b1) @ W2^T + b2) ----------------
// one wave per 16-row tile of one batch; grid = BB*128 blocks x 32 threads
__global__ void k_fstep(const bf16_t* __restrict__ m_bf,
                        const bf16_t* __restrict__ wl, const bf16_t* __restrict__ wr,
                        const float* __restrict__ b_f1,
                        const bf16_t* __restrict__ wf2, const float* __restrict__ b_f2,
                        float* __restrict__ F) {
    __shared__ bf16_t hid_lds[16 * LDS_STRIDE];
    int lane = threadIdx.x & 31;
    int b = blockIdx.x >> 7;
    int s0 = (blockIdx.x & 127) * 16;

    int lrow = s0 + (lane & 15);
    int rrow = lrow + 1; if (rrow > SS - 1) rrow = SS - 1;   // clamped halo
    const bf16_t* lptr = m_bf + (size_t)(b * SS + lrow) * DD;
    const bf16_t* rptr = m_bf + (size_t)(b * SS + rrow) * DD;

    v8f acc[8];
#pragma unroll
    for (int n = 0; n < 8; ++n)
#pragma unroll
        for (int i = 0; i < 8; ++i) acc[n][i] = 0.f;

#pragma unroll
    for (int k = 0; k < 4; ++k) {
        v16bf al = load_a_frag_rowptr(lptr, k * 32, lane);
        v16bf ar = load_a_frag_rowptr(rptr, k * 32, lane);
#pragma unroll
        for (int n = 0; n < 8; ++n) {
            v16bf bl = load_b_frag(wl, n * 16, k * 32, lane);
            acc[n] = wmma_bf16(al, bl, acc[n]);
            v16bf br = load_b_frag(wr, n * 16, k * 32, lane);
            acc[n] = wmma_bf16(ar, br, acc[n]);
        }
    }

    // hid = tanh(acc + b1) -> LDS (bf16, padded rows)
    int ncol = lane & 15;
    int rloc = (lane >> 4) << 3;
#pragma unroll
    for (int n = 0; n < 8; ++n) {
        float bias = b_f1[n * 16 + ncol];
#pragma unroll
        for (int v = 0; v < 8; ++v)
            hid_lds[(rloc + v) * LDS_STRIDE + n * 16 + ncol] = (bf16_t)tanhf(acc[n][v] + bias);
    }
    __syncthreads();

    v8f acc2[8];
#pragma unroll
    for (int n = 0; n < 8; ++n)
#pragma unroll
        for (int i = 0; i < 8; ++i) acc2[n][i] = 0.f;

    const bf16_t* hrow = hid_lds + (size_t)(lane & 15) * LDS_STRIDE;
#pragma unroll
    for (int k = 0; k < 4; ++k) {
        v16bf a = load_a_frag_rowptr(hrow, k * 32, lane);
#pragma unroll
        for (int n = 0; n < 8; ++n) {
            v16bf bfr = load_b_frag(wf2, n * 16, k * 32, lane);
            acc2[n] = wmma_bf16(a, bfr, acc2[n]);
        }
    }

#pragma unroll
    for (int n = 0; n < 8; ++n) {
        float bias = b_f2[n * 16 + ncol];
#pragma unroll
        for (int v = 0; v < 8; ++v) {
            int s = s0 + rloc + v;
            if (s < SS - 1)
                F[(size_t)(b * SS + s) * DD + n * 16 + ncol] = softplus_f(acc2[n][v] + bias);
        }
    }
}

// ---------------- m update: m = max(m + dt*(F[s-1]-F[s]), eps) ----------------
__global__ void k_update(float* __restrict__ m, bf16_t* __restrict__ m_bf,
                         const float* __restrict__ F, const float* __restrict__ cfl_raw) {
    int i = blockIdx.x * blockDim.x + threadIdx.x;
    if (i >= BB * SS * DD) return;
    int s = (i >> 7) & (SS - 1);
    float dt = 1.f / (1.f + __expf(-cfl_raw[0]));
    float Fm1 = (s >= 1)      ? F[i - DD] : 0.f;
    float Fs  = (s <= SS - 2) ? F[i]      : 0.f;
    float val = fmaxf(m[i] + dt * (Fm1 - Fs), EPSF);
    m[i] = val;
    m_bf[i] = (bf16_t)val;
}

// ---------------- decode: out = m @ w_dec^T + b_dec ----------------
// grid (250, 64), block 256 = 8 waves. Block tile: 128 rows x 128 cols.
// B tile (w_dec 128 cols x 128 k, bf16) staged in LDS once per block (async
// global->LDS when available); each wave preloads its 4 A fragments and feeds
// WMMAs from ds_load_b128.
__global__ void k_decode(const bf16_t* __restrict__ m_bf, const bf16_t* __restrict__ wdec_bf,
                         const float* __restrict__ b_dec, float* __restrict__ out) {
    __shared__ bf16_t btile[128 * LDS_STRIDE];   // ~34 KB
    int lane = threadIdx.x & 31;
    int wave = threadIdx.x >> 5;
    int n0_base = blockIdx.x * 128;
    int r0 = blockIdx.y * 128 + wave * 16;

    // cooperative stage of B tile: thread t copies 64 bf16 (row t/2, half t&1)
    {
        int c = threadIdx.x >> 1;
        int k0 = (threadIdx.x & 1) * 64;
        const bf16_t* src = wdec_bf + (size_t)(n0_base + c) * DD + k0;
        bf16_t* dst = btile + c * LDS_STRIDE + k0;
#ifdef USE_ASYNC_LDS
#pragma unroll
        for (int i = 0; i < 8; ++i)
            __builtin_amdgcn_global_load_async_to_lds_b128(
                (v4i_vs*)(src + i * 8),
                (v4i_vs*)(dst + i * 8),
                /*offset=*/0, /*cpol=*/0);
#else
#pragma unroll
        for (int i = 0; i < 8; ++i)
            *(v8bf*)(dst + i * 8) = *(const v8bf*)(src + i * 8);
#endif
    }

    // preload the 4 A fragments (reused across all 8 N subtiles)
    const bf16_t* arow = m_bf + (size_t)(r0 + (lane & 15)) * DD;
    v16bf afrag[4];
#pragma unroll
    for (int k = 0; k < 4; ++k) afrag[k] = load_a_frag_rowptr(arow, k * 32, lane);

#ifdef USE_ASYNC_LDS
# if __has_builtin(__builtin_amdgcn_s_wait_asynccnt)
    __builtin_amdgcn_s_wait_asynccnt(0);
# else
    asm volatile("s_wait_asynccnt 0x0" ::: "memory");
# endif
#endif
    __syncthreads();

    v8f acc[8];
#pragma unroll
    for (int n = 0; n < 8; ++n)
#pragma unroll
        for (int i = 0; i < 8; ++i) acc[n][i] = 0.f;

#pragma unroll
    for (int k = 0; k < 4; ++k) {
#pragma unroll
        for (int n = 0; n < 8; ++n) {
            v16bf b = load_b_frag_lds(btile, n * 16, k * 32, lane);
            acc[n] = wmma_bf16(afrag[k], b, acc[n]);
        }
    }

    int ncol = lane & 15;
    int rbase = r0 + ((lane >> 4) << 3);
#pragma unroll
    for (int n = 0; n < 8; ++n) {
        int col = n0_base + n * 16 + ncol;
        float bias = b_dec[col];
#pragma unroll
        for (int v = 0; v < 8; ++v)
            out[(size_t)(rbase + v) * VV + col] = acc[n][v] + bias;
    }
}

// ---------------- host launch ----------------
extern "C" void kernel_launch(void* const* d_in, const int* in_sizes, int n_in,
                              void* d_out, int out_size, void* d_ws, size_t ws_size,
                              hipStream_t stream) {
    const long long* x     = (const long long*)d_in[0];
    const float* emb       = (const float*)d_in[1];
    const float* w_mass    = (const float*)d_in[2];
    const float* b_mass    = (const float*)d_in[3];
    const float* w_f1      = (const float*)d_in[4];
    const float* b_f1      = (const float*)d_in[5];
    const float* w_f2      = (const float*)d_in[6];
    const float* b_f2      = (const float*)d_in[7];
    const float* cfl_raw   = (const float*)d_in[8];
    const float* w_dec     = (const float*)d_in[9];
    const float* b_dec     = (const float*)d_in[10];
    float* out = (float*)d_out;

    // workspace layout (bytes, 256-aligned)
    char* ws = (char*)d_ws;
    size_t off = 0;
    auto alloc = [&](size_t bytes) { char* p = ws + off; off = (off + bytes + 255) & ~(size_t)255; return p; };
    float*  m_f32    = (float*)alloc((size_t)BB * SS * DD * 4);
    bf16_t* m_bf     = (bf16_t*)alloc((size_t)BB * SS * DD * 2);
    float*  F_f32    = (float*)alloc((size_t)BB * SS * DD * 4);
    bf16_t* wdec_bf  = (bf16_t*)alloc((size_t)VV * DD * 2);
    bf16_t* wmass_bf = (bf16_t*)alloc((size_t)DD * DD * 2);
    bf16_t* wl_bf    = (bf16_t*)alloc((size_t)DD * DD * 2);
    bf16_t* wr_bf    = (bf16_t*)alloc((size_t)DD * DD * 2);
    bf16_t* wf2_bf   = (bf16_t*)alloc((size_t)DD * DD * 2);

    // 1) weight conversions to bf16
    {
        int n = VV * DD;
        k_cvt_bf16<<<(n + 255) / 256, 256, 0, stream>>>(w_dec, wdec_bf, n);
    }
    {
        int n = DD * DD;
        k_cvt_bf16<<<(n + 255) / 256, 256, 0, stream>>>(w_mass, wmass_bf, n);
        k_cvt_bf16<<<(n + 255) / 256, 256, 0, stream>>>(w_f2, wf2_bf, n);
        k_split_wf1<<<(n + 255) / 256, 256, 0, stream>>>(w_f1, wl_bf, wr_bf);
    }

    // 2) embedding + mass head
    k_embed_mass<<<(BB * SS) / 16, 32, 0, stream>>>(x, emb, wmass_bf, b_mass, m_f32, m_bf);

    // 3) K flux iterations
    for (int k = 0; k < KITER; ++k) {
        k_fstep<<<BB * (SS / 16), 32, 0, stream>>>(m_bf, wl_bf, wr_bf, b_f1, wf2_bf, b_f2, F_f32);
        int n = BB * SS * DD;
        k_update<<<(n + 255) / 256, 256, 0, stream>>>(m_f32, m_bf, F_f32, cfl_raw);
    }

    // 4) decode to vocab
    k_decode<<<dim3(VV / 128, (BB * SS) / 128), 256, 0, stream>>>(m_bf, wdec_bf, b_dec, out);
}
